// Attention_87058987090007
// MI455X (gfx1250) — matrix operand order repaired and checked by reading, not compiled, not run on hardware
//
#include <hip/hip_runtime.h>

typedef _Float16 f16_t;
typedef __attribute__((ext_vector_type(16))) _Float16 v16h;
typedef __attribute__((ext_vector_type(8)))  _Float16 v8h;
typedef __attribute__((ext_vector_type(8)))  float    v8f;
typedef __attribute__((ext_vector_type(4)))  float    v4f;

#define D_MODEL  768
#define NSEQ     1024
#define BATCH    8
#define NHEAD    12
#define DHEAD    64
#define QKV_COLS 2304   // 3 * D_MODEL

union frag16 { v16h v; v8h h[2]; };

// ---------------------------------------------------------------------------
// Fragment loaders. ISA 7.12.2 layouts are K-contiguous per half-wave:
//   A (16x32, M=lane&15): values = p[8*hh .. +7] ++ p[16+8*hh .. +7]
//   B (32x16, N=lane&15): values = col[8*hh .. +7] ++ col[16+8*hh .. +7]
// With the right storage layout each fragment is exactly two b128 loads.
// ---------------------------------------------------------------------------
__device__ __forceinline__ v16h load_a_f16(const f16_t* p /*row base + kbase*/, int hh) {
    frag16 f;
    f.h[0] = *(const v8h*)(p + 8 * hh);
    f.h[1] = *(const v8h*)(p + 16 + 8 * hh);
    return f.v;
}

__device__ __forceinline__ v16h load_b_f16(const f16_t* p /*col base + kbase*/, int hh) {
    frag16 f;
    f.h[0] = *(const v8h*)(p + 8 * hh);
    f.h[1] = *(const v8h*)(p + 16 + 8 * hh);
    return f.v;
}

// A fragment from f32 source: four float4 loads + packed in-register cvt
__device__ __forceinline__ v16h load_a_f32cvt(const float* p, int hh) {
    v4f x0 = *(const v4f*)(p + 8 * hh);
    v4f x1 = *(const v4f*)(p + 8 * hh + 4);
    v4f y0 = *(const v4f*)(p + 16 + 8 * hh);
    v4f y1 = *(const v4f*)(p + 16 + 8 * hh + 4);
    frag16 f;
#pragma unroll
    for (int i = 0; i < 4; ++i) {
        f.v[i]      = (f16_t)x0[i];
        f.v[4 + i]  = (f16_t)x1[i];
        f.v[8 + i]  = (f16_t)y0[i];
        f.v[12 + i] = (f16_t)y1[i];
    }
    return f.v;
}

// ---------------------------------------------------------------------------
// CDNA5 async copy: per-lane 16B global -> LDS, tracked by ASYNCcnt.
// lds_off is a byte offset within this workgroup's LDS allocation.
// ---------------------------------------------------------------------------
__device__ __forceinline__ void async_g2l_b128(unsigned lds_off, const void* gaddr) {
    asm volatile("global_load_async_to_lds_b128 %0, %1, off"
                 :: "v"(lds_off), "v"(gaddr) : "memory");
}

// Fetch one 64x32 f16 V-tile (4KB) into LDS: 8 chunks of 16B per lane.
__device__ __forceinline__ void vtile_fetch(const f16_t* __restrict__ Vth, int kb,
                                            unsigned lds_off, int lane) {
#pragma unroll
    for (int j = 0; j < 8; ++j) {
        int idx   = j * 32 + lane;     // 256 16B chunks
        int row   = idx >> 2;          // d index 0..63
        int chunk = idx & 3;           // 16B chunk within 64B row
        async_g2l_b128(lds_off + (unsigned)(row * 64 + chunk * 16),
                       (const void*)(Vth + (size_t)row * NSEQ + kb + chunk * 8));
    }
}

// ---------------------------------------------------------------------------
// Convert f32 weight [rows x cols] -> f16 TRANSPOSED [cols x rows]
// (one-time; makes every GEMM B-fragment two b128 loads)
// ---------------------------------------------------------------------------
__global__ void cvt_transpose_f16(const float* __restrict__ in, f16_t* __restrict__ out,
                                  int rows, int cols) {
    int i = blockIdx.x * blockDim.x + threadIdx.x;
    int n = rows * cols;
    if (i < n) {
        int r = i / cols, c = i % cols;
        out[(size_t)c * rows + r] = (f16_t)in[i];
    }
}

// ---------------------------------------------------------------------------
// QKV GEMM: [8192 x 768] f32 @ Wt[2304 x 768] f16 -> Q/K f16 [B,H,N,d],
//           V f16 TRANSPOSED [B,H,d,N].
// grid (36, 256): x = 64-col tile (exactly one head of q|k|v), y = 32-row tile.
// ---------------------------------------------------------------------------
__global__ void qkv_gemm(const float* __restrict__ x, const f16_t* __restrict__ Wt,
                         f16_t* __restrict__ q, f16_t* __restrict__ k, f16_t* __restrict__ v) {
    const int T        = blockIdx.x;            // 0..35
    const int row_base = blockIdx.y * 32;       // row over B*N
    const int col_base = T * 64;
    const int which    = T / 12;                // 0=q 1=k 2=v
    const int head     = T % 12;

    const int m  = threadIdx.x & 15;
    const int hh = threadIdx.x >> 4;

    v8f c[2][4] = {};
    for (int kb = 0; kb < D_MODEL; kb += 32) {
        v16h a0 = load_a_f32cvt(x + (size_t)(row_base + m) * D_MODEL + kb, hh);
        v16h a1 = load_a_f32cvt(x + (size_t)(row_base + 16 + m) * D_MODEL + kb, hh);
#pragma unroll
        for (int t = 0; t < 4; ++t) {
            // Wt is [2304 x 768]: column (col_base+16t+m) of W is a contiguous row
            v16h b = load_b_f16(Wt + (size_t)(col_base + 16 * t + m) * D_MODEL + kb, hh);
            c[0][t] = __builtin_amdgcn_wmma_f32_16x16x32_f16(false, a0, false, b,
                                                             (short)0, c[0][t], false, false);
            c[1][t] = __builtin_amdgcn_wmma_f32_16x16x32_f16(false, a1, false, b,
                                                             (short)0, c[1][t], false, false);
        }
    }

#pragma unroll
    for (int aa = 0; aa < 2; ++aa) {
#pragma unroll
        for (int t = 0; t < 4; ++t) {
#pragma unroll
            for (int r = 0; r < 8; ++r) {
                int row  = r + 8 * hh;               // C/D layout: M = r + 8*half
                int grow = row_base + 16 * aa + row; // b*1024 + n
                int b_   = grow >> 10;
                int n_   = grow & 1023;
                int dd   = 16 * t + m;
                f16_t val = (f16_t)c[aa][t][r];
                if (which == 0)
                    q[(((size_t)b_ * NHEAD + head) * NSEQ + n_) * DHEAD + dd] = val;
                else if (which == 1)
                    k[(((size_t)b_ * NHEAD + head) * NSEQ + n_) * DHEAD + dd] = val;
                else // V stored transposed: [b,h][dd][n]
                    v[(((size_t)b_ * NHEAD + head) * DHEAD + dd) * NSEQ + n_] = val;
            }
        }
    }
}

// ---------------------------------------------------------------------------
// Attention: one wave per (b, h, 16-query tile). grid (64, 96): x=qt, y=bh.
//   Phase 1: S[16x1024] = scale * Q Ktile^T  (WMMA, f32 into LDS)
//   Phase 2: softmax over LDS rows (trans-pipe exp), P f16 into LDS
//   Phase 3: O = P @ V with V tiles double-buffered in LDS via
//            global_load_async_to_lds_b128 / s_wait_asynccnt pipelining.
// ---------------------------------------------------------------------------
__global__ void attn_kernel(const f16_t* __restrict__ Q, const f16_t* __restrict__ K,
                            const f16_t* __restrict__ Vt, f16_t* __restrict__ O) {
    extern __shared__ __align__(16) char smem[];
    float* S      = (float*)smem;                                   // 16*1024 f32 (64KB)
    f16_t* P      = (f16_t*)(smem + 16 * NSEQ * sizeof(float));     // 16*1024 f16 (32KB)
    float* redmax = (float*)(P + 16 * NSEQ);                        // [32]
    float* rowmax = redmax + 32;                                    // [16]
    float* redsum = rowmax + 16;                                    // [32]
    float* rowinv = redsum + 32;                                    // [16]
    f16_t* Vl     = (f16_t*)(rowinv + 16);                          // [2][64][32] f16 (8KB)

    const int qt     = blockIdx.x;           // 0..63
    const int bh     = blockIdx.y;           // b*NHEAD + h
    const int q_base = qt * 16;
    const f16_t* Qh  = Q  + (size_t)bh * NSEQ * DHEAD;
    const f16_t* Kh  = K  + (size_t)bh * NSEQ * DHEAD;
    const f16_t* Vth = Vt + (size_t)bh * DHEAD * NSEQ;   // [d][n]

    const int lane = threadIdx.x;
    const int m  = lane & 15;
    const int hh = lane >> 4;
    const float scale = 0.125f;              // d^-0.5, d=64

    // Preload Q tile 16x64 as two A fragments (K=0..31, 32..63)
    v16h aq[2];
#pragma unroll
    for (int s = 0; s < 2; ++s)
        aq[s] = load_a_f16(Qh + (size_t)(q_base + m) * DHEAD + 32 * s, hh);

    // ---- Phase 1: scores into LDS ----
    for (int kt = 0; kt < NSEQ / 16; ++kt) {
        v8f c = {};
#pragma unroll
        for (int s = 0; s < 2; ++s) {
            // B = K^T: column n is key kt*16+n; its K-dim (d) is contiguous in Kh
            v16h b = load_b_f16(Kh + (size_t)(kt * 16 + m) * DHEAD + 32 * s, hh);
            c = __builtin_amdgcn_wmma_f32_16x16x32_f16(false, aq[s], false, b,
                                                       (short)0, c, false, false);
        }
#pragma unroll
        for (int r = 0; r < 8; ++r)
            S[(r + 8 * hh) * NSEQ + kt * 16 + m] = c[r] * scale;
    }
    __syncthreads();

    // ---- Phase 2: softmax (each thread owns half of one row) ----
    {
        const int   row  = lane & 15;
        const int   side = lane >> 4;
        const float* sr  = S + row * NSEQ + side * (NSEQ / 2);
        float mx = -3.0e38f;
        for (int i = 0; i < NSEQ / 2; ++i) mx = fmaxf(mx, sr[i]);
        redmax[lane] = mx;
        __syncthreads();
        if (lane < 16)
            rowmax[lane] = fmaxf(redmax[lane], redmax[lane + 16]);
        __syncthreads();
        const float rm = rowmax[row];
        f16_t* pr = P + row * NSEQ + side * (NSEQ / 2);
        float sum = 0.0f;
        for (int i = 0; i < NSEQ / 2; ++i) {
            float e = __expf(sr[i] - rm);
            pr[i] = (f16_t)e;                 // unnormalized; normalize at store
            sum += e;
        }
        redsum[lane] = sum;
        __syncthreads();
        if (lane < 16)
            rowinv[lane] = 1.0f / (redsum[lane] + redsum[lane + 16]);
        __syncthreads();
    }

    // ---- Phase 3: O = P @ V, async double-buffered V tiles ----
    const unsigned vl_off = (unsigned)((char*)Vl - smem);  // dynamic LDS starts at 0
    vtile_fetch(Vth, 0, vl_off, lane);                     // prefetch tile 0

    v8f o[4] = {};
    for (int kb = 0; kb < NSEQ; kb += 32) {
        const int cur = (kb >> 5) & 1;
        if (kb + 32 < NSEQ) {
            // WAR guard: previous iteration's ds_load_b128 reads of buffer cur^1
            asm volatile("s_wait_dscnt 0x0" ::: "memory");
            vtile_fetch(Vth, kb + 32, vl_off + (unsigned)((cur ^ 1) * 4096), lane);
            asm volatile("s_wait_asynccnt 0x8" ::: "memory");  // tile `cur` landed
        } else {
            asm volatile("s_wait_asynccnt 0x0" ::: "memory");
        }
        const f16_t* Vcur = Vl + cur * 2048;               // [64][32] f16
        v16h a = load_a_f16(P + m * NSEQ + kb, hh);        // LDS b128 reads
#pragma unroll
        for (int t = 0; t < 4; ++t) {
            // B(kk, n) = Vtile[dcol=16t+n][kk]: K-contiguous in the LDS tile
            v16h b = load_b_f16(Vcur + (16 * t + m) * 32, hh);
            o[t] = __builtin_amdgcn_wmma_f32_16x16x32_f16(false, a, false, b,
                                                          (short)0, o[t], false, false);
        }
    }

    const int b_ = bh / NHEAD, head = bh % NHEAD;
#pragma unroll
    for (int t = 0; t < 4; ++t) {
#pragma unroll
        for (int r = 0; r < 8; ++r) {
            int row = r + 8 * hh;
            float val = o[t][r] * rowinv[row];
            O[(size_t)(b_ * NSEQ + q_base + row) * D_MODEL + head * DHEAD + 16 * t + m] = (f16_t)val;
        }
    }
}

// ---------------------------------------------------------------------------
// Proj GEMM: [8192 x 768] f16 @ Wt[768 x 768] f16 + bias -> f32 output
// grid (12, 256): x = 64-col tile, y = 32-row tile.
// ---------------------------------------------------------------------------
__global__ void proj_gemm(const f16_t* __restrict__ A, const f16_t* __restrict__ Wt,
                          const float* __restrict__ bias, float* __restrict__ out) {
    const int row_base = blockIdx.y * 32;
    const int col_base = blockIdx.x * 64;
    const int m  = threadIdx.x & 15;
    const int hh = threadIdx.x >> 4;

    v8f c[2][4] = {};
    for (int kb = 0; kb < D_MODEL; kb += 32) {
        v16h a0 = load_a_f16(A + (size_t)(row_base + m) * D_MODEL + kb, hh);
        v16h a1 = load_a_f16(A + (size_t)(row_base + 16 + m) * D_MODEL + kb, hh);
#pragma unroll
        for (int t = 0; t < 4; ++t) {
            v16h b = load_b_f16(Wt + (size_t)(col_base + 16 * t + m) * D_MODEL + kb, hh);
            c[0][t] = __builtin_amdgcn_wmma_f32_16x16x32_f16(false, a0, false, b,
                                                             (short)0, c[0][t], false, false);
            c[1][t] = __builtin_amdgcn_wmma_f32_16x16x32_f16(false, a1, false, b,
                                                             (short)0, c[1][t], false, false);
        }
    }
#pragma unroll
    for (int aa = 0; aa < 2; ++aa) {
#pragma unroll
        for (int t = 0; t < 4; ++t) {
#pragma unroll
            for (int r = 0; r < 8; ++r) {
                int row = 16 * aa + r + 8 * hh;
                int col = col_base + 16 * t + m;
                out[(size_t)(row_base + row) * D_MODEL + col] = c[aa][t][r] + bias[col];
            }
        }
    }
}

// ---------------------------------------------------------------------------
extern "C" void kernel_launch(void* const* d_in, const int* in_sizes, int n_in,
                              void* d_out, int out_size, void* d_ws, size_t ws_size,
                              hipStream_t stream) {
    const float* x     = (const float*)d_in[0];   // [8,1024,768]
    const float* Wqkv  = (const float*)d_in[1];   // [768,2304]
    const float* Wproj = (const float*)d_in[2];   // [768,768]
    const float* bproj = (const float*)d_in[3];   // [768]
    float* out = (float*)d_out;                   // [8,1024,768]

    const size_t nWqkv  = (size_t)D_MODEL * QKV_COLS;           // 1,769,472
    const size_t nWproj = (size_t)D_MODEL * D_MODEL;            //   589,824
    const size_t nHead  = (size_t)BATCH * NHEAD * NSEQ * DHEAD; // 6,291,456

    f16_t* WqkvT  = (f16_t*)d_ws;           // [2304 x 768]
    f16_t* WprojT = WqkvT + nWqkv;          // [768 x 768]
    f16_t* qbuf   = WprojT + nWproj;        // [B,H,N,d]
    f16_t* kbuf   = qbuf + nHead;           // [B,H,N,d]
    f16_t* vbuf   = kbuf + nHead;           // [B,H,d,N]  (transposed)
    f16_t* attnb  = vbuf + nHead;           // [B*N, 768]
    // total ws: ~52.5 MB of f16

    cvt_transpose_f16<<<(int)((nWqkv  + 255) / 256), 256, 0, stream>>>(Wqkv,  WqkvT,  D_MODEL, QKV_COLS);
    cvt_transpose_f16<<<(int)((nWproj + 255) / 256), 256, 0, stream>>>(Wproj, WprojT, D_MODEL, D_MODEL);

    // QKV GEMM: col-tile x row-tile grid, 1 wave each
    qkv_gemm<<<dim3(36, 256), 32, 0, stream>>>(x, WqkvT, qbuf, kbuf, vbuf);

    // Attention: LDS = 64KB scores + 32KB probs + stats + 8KB V double-buffer
    const size_t smem = 16 * NSEQ * sizeof(float) + 16 * NSEQ * sizeof(f16_t)
                      + (32 + 16 + 32 + 16) * sizeof(float)
                      + 2 * 64 * 32 * sizeof(f16_t);
    attn_kernel<<<dim3(NSEQ / 16, BATCH * NHEAD), 32, smem, stream>>>(qbuf, kbuf, vbuf, attnb);

    // Proj GEMM
    proj_gemm<<<dim3(12, 256), 32, 0, stream>>>(attnb, WprojT, bproj, out);
}
